// ThreeLayers_72155450573454
// MI455X (gfx1250) — matrix operand order repaired and checked by reading, not compiled, not run on hardware
//
#include <hip/hip_runtime.h>
#include <hip/hip_bf16.h>

// ---------------------------------------------------------------------------
// Hetero-GAT (3 relations, 2 layers) for MI455X / gfx1250.
//  - Linear maps: bf16 WMMA (v_wmma_f32_16x16x32_bf16), f32 accumulate.
//    X converted to bf16 once per layer; W converted + transposed once per
//    relation -> all WMMA fragments load as contiguous global_load_b128.
//  - Attention: segment softmax seeded by self-loop, float atomics in L2
//    (acc is <=51MB, fits the 192MB L2, so scatter atomics stay on-chip).
// ---------------------------------------------------------------------------

#define N_NODES 50000
#define N_EDGES 400000
#define IN_F    128
#define HID     128
#define EMB     256
#define KIN     128   // inner dim of both layers' GEMMs (IN_F == HID == 128)

typedef __attribute__((ext_vector_type(16))) __bf16 v16bf;
typedef __attribute__((ext_vector_type(8)))  __bf16 v8bf;
typedef __attribute__((ext_vector_type(8)))  float  v8f;

__device__ __forceinline__ __bf16 f2bf(float f) {
  unsigned u = __builtin_bit_cast(unsigned, f);
  unsigned r = (u + 0x7FFFu + ((u >> 16) & 1u)) >> 16;      // round-to-nearest-even
  unsigned short s = (unsigned short)r;
  return __builtin_bit_cast(__bf16, s);
}

__device__ __forceinline__ float leaky(float x) { return x > 0.f ? x : 0.2f * x; }

__device__ __forceinline__ void atomicMaxF(float* addr, float val) {
  // monotone-toward-max float atomic via integer ordering trick
  if (val >= 0.f) atomicMax((int*)addr, __float_as_int(val));
  else            atomicMin((unsigned int*)addr, __float_as_uint(val));
}

// --------------------------- WMMA GEMM -------------------------------------
// C[N x kout] = Xb[N x 128] * Wt^T, with Wt stored transposed: Wt[col*128+k].
// One wave per 16x16 C tile, K fully unrolled (4 x v_wmma_f32_16x16x32_bf16).
// A frag (16-bit 16x32): lane<16 -> M=lane, K=kk..kk+15;
//                        lane>=16 -> same M, K shifted by +16.
// B frag (32x16): lane&15 = N col; lanes<16 hold K kk..kk+15, lanes>=16 +16.
__global__ void gemm_bf16_wmma(const __bf16* __restrict__ Xb,
                               const __bf16* __restrict__ Wt,
                               float* __restrict__ C, int kout) {
  const int lane    = threadIdx.x & 31;
  const int halfsel = lane >> 4;          // 0 or 1
  const int l15     = lane & 15;
  const int tr      = blockIdx.x;
  const int tc      = blockIdx.y;
  const __bf16* xrow = Xb + (long)(tr * 16 + l15) * KIN;           // A: row M
  const __bf16* wrow = Wt + (long)(tc * 16 + l15) * KIN;           // B: col N

  v8f acc = {};
#pragma unroll
  for (int kk = 0; kk < KIN; kk += 32) {
    const int ka = kk + halfsel * 16;      // A: one contiguous 16-elem run
    v8bf alo = *(const v8bf*)(xrow + ka);
    v8bf ahi = *(const v8bf*)(xrow + ka + 8);
    v16bf a = __builtin_shufflevector(alo, ahi, 0,1,2,3,4,5,6,7,
                                                8,9,10,11,12,13,14,15);
    const int kb = kk + halfsel * 16;      // B: one contiguous 16-elem run
    v8bf blo = *(const v8bf*)(wrow + kb);
    v8bf bhi = *(const v8bf*)(wrow + kb + 8);
    v16bf b = __builtin_shufflevector(blo, bhi, 0,1,2,3,4,5,6,7,
                                                8,9,10,11,12,13,14,15);
    acc = __builtin_amdgcn_wmma_f32_16x16x32_bf16(false, a, false, b,
                                                  (short)0, acc, false, false);
  }
  // D layout: VGPR i -> row M = i + 8*halfsel, col N = l15
  float* crow = C + (long)(tr * 16 + halfsel * 8) * kout + tc * 16 + l15;
#pragma unroll
  for (int i = 0; i < 8; ++i)
    crow[(long)i * kout] = acc[i];
}

// --------------------------- helpers ---------------------------------------
// plain f32 -> bf16 (used for X, once per layer)
__global__ void convert_bf16_kernel(const float* __restrict__ w,
                                    __bf16* __restrict__ wb, long n) {
  long i = (long)blockIdx.x * blockDim.x + threadIdx.x;
  if (i < n) wb[i] = f2bf(w[i]);
}

// f32 W[kin x kout] -> bf16 transposed Wt[kout x kin]
__global__ void convert_wt_kernel(const float* __restrict__ w,
                                  __bf16* __restrict__ wt, int kin, int kout) {
  int i = blockIdx.x * blockDim.x + threadIdx.x;
  if (i >= kin * kout) return;
  int k = i / kout, c = i % kout;
  wt[(long)c * kin + k] = f2bf(w[i]);
}

__global__ void zero_kernel(float* __restrict__ p, long n) {
  long i = (long)blockIdx.x * blockDim.x + threadIdx.x;
  if (i < n) p[i] = 0.f;
}

// es[i] = dot(H[i,:], a[:]); one wave per node
__global__ void score_kernel(const float* __restrict__ H,
                             const float* __restrict__ a,
                             float* __restrict__ out, int n, int k) {
  int w    = (int)(((long)blockIdx.x * blockDim.x + threadIdx.x) >> 5);
  int lane = threadIdx.x & 31;
  if (w >= n) return;
  const float* row = H + (long)w * k;
  float s = 0.f;
  for (int i = lane; i < k; i += 32) s += row[i] * a[i];
#pragma unroll
  for (int m = 16; m; m >>= 1) s += __shfl_xor(s, m, 32);
  if (lane == 0) out[w] = s;
}

// m[i] = leaky(es[i]+ed[i])  (self loop guarantees a finite max seed)
__global__ void init_m_kernel(const float* __restrict__ es,
                              const float* __restrict__ ed,
                              float* __restrict__ m, int n) {
  int i = blockIdx.x * blockDim.x + threadIdx.x;
  if (i < n) m[i] = leaky(es[i] + ed[i]);
}

__global__ void edge_max_kernel(const int* __restrict__ src, const int* __restrict__ dst,
                                const float* __restrict__ es, const float* __restrict__ ed,
                                float* __restrict__ m, int ne) {
  int e = blockIdx.x * blockDim.x + threadIdx.x;
  if (e >= ne) return;
  int s = src[e], d = dst[e];
  if (s != d) atomicMaxF(&m[d], leaky(es[s] + ed[d]));
}

// sb[i] = exp(alpha_self - m[i])   (self-loop term of the softmax denominator)
__global__ void node_exp_kernel(const float* __restrict__ es, const float* __restrict__ ed,
                                const float* __restrict__ m, float* __restrict__ sb, int n) {
  int i = blockIdx.x * blockDim.x + threadIdx.x;
  if (i < n) sb[i] = __expf(leaky(es[i] + ed[i]) - m[i]);
}

__global__ void edge_sum_kernel(const int* __restrict__ src, const int* __restrict__ dst,
                                const float* __restrict__ es, const float* __restrict__ ed,
                                const float* __restrict__ m, float* __restrict__ sb, int ne) {
  int e = blockIdx.x * blockDim.x + threadIdx.x;
  if (e >= ne) return;
  int s = src[e], d = dst[e];
  if (s != d) atomicAdd(&sb[d], __expf(leaky(es[s] + ed[d]) - m[d]));
}

// self-loop message + per-relation bias: acc[i,:] += w_self*hs[i,:] + b[:]
__global__ void self_agg_kernel(const float* __restrict__ hs,
                                const float* __restrict__ es, const float* __restrict__ ed,
                                const float* __restrict__ m, const float* __restrict__ sb,
                                const float* __restrict__ bias,
                                float* __restrict__ acc, int k) {
  int i = blockIdx.x;          // node
  int t = threadIdx.x;         // feature (blockDim == k)
  float w = __expf(leaky(es[i] + ed[i]) - m[i]) / sb[i];
  long idx = (long)i * k + t;
  acc[idx] += w * hs[idx] + bias[t];
}

// edge messages: one wave per edge, vector scatter-add into L2-resident acc
__global__ void edge_agg_kernel(const int* __restrict__ src, const int* __restrict__ dst,
                                const float* __restrict__ es, const float* __restrict__ ed,
                                const float* __restrict__ m, const float* __restrict__ sb,
                                const float* __restrict__ hs,
                                float* __restrict__ acc, int ne, int k) {
  int e    = (int)(((long)blockIdx.x * blockDim.x + threadIdx.x) >> 5);
  int lane = threadIdx.x & 31;
  if (e >= ne) return;
  int s = src[e], d = dst[e];
  if (s == d) return;
  float w = __expf(leaky(es[s] + ed[d]) - m[d]) / sb[d];
  const float* hrow = hs + (long)s * k;
  float*       arow = acc + (long)d * k;
  for (int j = lane; j < k; j += 32) atomicAdd(&arow[j], w * hrow[j]);
}

// out = acc / 3 (HeteroConv aggr='mean' over 3 relations)
__global__ void mean_kernel(const float* __restrict__ acc, float* __restrict__ out, long n) {
  long i = (long)blockIdx.x * blockDim.x + threadIdx.x;
  if (i < n) out[i] = acc[i] * (1.0f / 3.0f);
}

// --------------------------- host-side layer driver ------------------------
static void run_layer(const float* X, int kout,
                      const float* Ws, const float* Wd,
                      const float* as_, const float* ad_, const float* bias,
                      const int* esrc, const int* edst,
                      __bf16* xb, __bf16* wb,
                      float* hs, float* hd, float* es, float* ed,
                      float* m, float* sb,
                      float* acc, float* out, hipStream_t stream) {
  const int n = N_NODES, ne = N_EDGES;
  const int nw = KIN * kout;
  dim3 gGemm((n + 15) / 16, kout / 16), bGemm(32);
  const int TB  = 256;
  const int gN  = (n  + TB - 1) / TB;
  const int gE  = (ne + TB - 1) / TB;
  const int gNW = (n * 32 + TB - 1) / TB;                     // one wave per node
  const int gEW = (int)(((long)ne * 32 + TB - 1) / TB);       // one wave per edge

  // X -> bf16 once per layer
  convert_bf16_kernel<<<(int)(((long)n * KIN + TB - 1) / TB), TB, 0, stream>>>(
      X, xb, (long)n * KIN);
  zero_kernel<<<(int)(((long)n * kout + TB - 1) / TB), TB, 0, stream>>>(acc, (long)n * kout);

  for (int r = 0; r < 3; ++r) {
    // hs = X @ Ws[r]; hd = X @ Wd[r]   (bf16 WMMA, transposed bf16 weights)
    convert_wt_kernel<<<(nw + TB - 1) / TB, TB, 0, stream>>>(Ws + (long)r * nw, wb, KIN, kout);
    gemm_bf16_wmma<<<gGemm, bGemm, 0, stream>>>(xb, wb, hs, kout);
    convert_wt_kernel<<<(nw + TB - 1) / TB, TB, 0, stream>>>(Wd + (long)r * nw, wb, KIN, kout);
    gemm_bf16_wmma<<<gGemm, bGemm, 0, stream>>>(xb, wb, hd, kout);
    // attention scores
    score_kernel<<<gNW, TB, 0, stream>>>(hs, as_ + (long)r * kout, es, n, kout);
    score_kernel<<<gNW, TB, 0, stream>>>(hd, ad_ + (long)r * kout, ed, n, kout);
    // segment softmax (self-loop seeds the max; edges refine it)
    init_m_kernel<<<gN, TB, 0, stream>>>(es, ed, m, n);
    edge_max_kernel<<<gE, TB, 0, stream>>>(esrc + (long)r * ne, edst + (long)r * ne, es, ed, m, ne);
    node_exp_kernel<<<gN, TB, 0, stream>>>(es, ed, m, sb, n);
    edge_sum_kernel<<<gE, TB, 0, stream>>>(esrc + (long)r * ne, edst + (long)r * ne, es, ed, m, sb, ne);
    // weighted aggregation
    self_agg_kernel<<<n, kout, 0, stream>>>(hs, es, ed, m, sb, bias + (long)r * kout, acc, kout);
    edge_agg_kernel<<<gEW, TB, 0, stream>>>(esrc + (long)r * ne, edst + (long)r * ne,
                                            es, ed, m, sb, hs, acc, ne, kout);
  }
  mean_kernel<<<(int)(((long)n * kout + TB - 1) / TB), TB, 0, stream>>>(acc, out, (long)n * kout);
}

extern "C" void kernel_launch(void* const* d_in, const int* in_sizes, int n_in,
                              void* d_out, int out_size, void* d_ws, size_t ws_size,
                              hipStream_t stream) {
  const float* x    = (const float*)d_in[0];
  const int*   src  = (const int*)  d_in[1];   // [3, E]
  const int*   dst  = (const int*)  d_in[2];   // [3, E]
  const float* Ws1  = (const float*)d_in[3];
  const float* Wd1  = (const float*)d_in[4];
  const float* as1  = (const float*)d_in[5];
  const float* ad1  = (const float*)d_in[6];
  const float* b1   = (const float*)d_in[7];
  const float* Ws2  = (const float*)d_in[8];
  const float* Wd2  = (const float*)d_in[9];
  const float* as2  = (const float*)d_in[10];
  const float* ad2  = (const float*)d_in[11];
  const float* b2   = (const float*)d_in[12];
  float* out = (float*)d_out;

  // scratch carve-up (float units)
  float* ws  = (float*)d_ws;
  float* h1  = ws;                       ws += (long)N_NODES * HID;   // layer-1 output
  float* hs  = ws;                       ws += (long)N_NODES * EMB;
  float* hd  = ws;                       ws += (long)N_NODES * EMB;
  float* acc = ws;                       ws += (long)N_NODES * EMB;
  float* es  = ws;                       ws += N_NODES;
  float* ed  = ws;                       ws += N_NODES;
  float* m   = ws;                       ws += N_NODES;
  float* sb  = ws;                       ws += N_NODES;
  __bf16* xb = (__bf16*)ws;              ws += (long)N_NODES * KIN / 2;  // bf16 X
  __bf16* wb = (__bf16*)ws;              // KIN*EMB bf16 = 64 KB (transposed W)

  // layer 1: [N,128] -> [N,128]
  run_layer(x, HID, Ws1, Wd1, as1, ad1, b1, src, dst,
            xb, wb, hs, hd, es, ed, m, sb, acc, h1, stream);
  // layer 2: [N,128] -> [N,256]
  run_layer(h1, EMB, Ws2, Wd2, as2, ad2, b2, src, dst,
            xb, wb, hs, hd, es, ed, m, sb, acc, out, stream);
  (void)in_sizes; (void)n_in; (void)out_size; (void)ws_size;
}
